// TriangleAttention_57561151701398
// MI455X (gfx1250) — matrix operand order repaired and checked by reading, not compiled, not run on hardware
//
#include <hip/hip_runtime.h>
#include <hip/hip_bf16.h>

typedef __attribute__((ext_vector_type(16))) _Float16 v16h;
typedef __attribute__((ext_vector_type(8)))  float    v8f;

#define C_IN   128
#define C_HID  128
#define NH     4
#define HD     32
#define I_DIM  256
#define J_DIM  256
#define NROW   (I_DIM * J_DIM)      // 65536
#define EPS_LN 1e-5f
#define BIGINF 1e9f
#define QSCALE 0.17677669529663687f // 1/sqrt(32)

// ---------------- workspace layout (bytes) ----------------
#define OFF_XN    0ull                                    // fp16 [65536][128]
#define OFF_Q     (OFF_XN   + (size_t)NROW * C_HID * 2)   // fp16 [I][H][J][HD]
#define OFF_K     (OFF_Q    + (size_t)NROW * C_HID * 2)
#define OFF_V     (OFF_K    + (size_t)NROW * C_HID * 2)
#define OFF_OG    (OFF_V    + (size_t)NROW * C_HID * 2)   // fp16 [65536][128]: g, then o*g
#define OFF_TRI   (OFF_OG   + (size_t)NROW * C_HID * 2)   // f32  [H][65536]
#define OFF_WQKVG (OFF_TRI  + (size_t)NH * NROW * 4)      // fp16 [512][128]
#define OFF_WO    (OFF_WQKVG + (size_t)512 * 128 * 2)     // fp16 [128][128]

// k_attn dynamic LDS layout (bytes within dynamic region)
#define LDS_VT    0                      // [HD][256] f16 transposed V, 16 KB
#define LDS_KS    16384                  // [256][HD] f16 K rows, 16 KB
#define LDS_VROW  32768                  // [256][HD] f16 V rows (async-staged), 16 KB
#define LDS_PBASE 49152                  // 8 x [16][256] f16 prob strips, 64 KB
#define LDS_TOTAL 114688                 // 112 KB < 320 KB/WGP

// ---------------- WMMA fragment helpers (ISA 7.12.2 layouts) ----------------
// A 16x32 f16: lane L -> row (L&15); VGPR0..3 hold K = (L>>4)*8 + 0..7,
//              VGPR4..7 hold K = 16 + (L>>4)*8 + 0..7.
__device__ __forceinline__ v16h load_a16x32(const _Float16* base, int ld, int lane) {
  union { v16h h; uint4 u[2]; } f;
  const _Float16* p = base + (size_t)(lane & 15) * ld + (lane >> 4) * 8;
  f.u[0] = *(const uint4*)(p);
  f.u[1] = *(const uint4*)(p + 16);
  return f.h;
}
// B 32x16 f16, element (K=k, N=n) stored N-major at base[n*ld + k]:
// lane L -> col (L&15); halves 0..7 hold K = (L>>4)*8 + 0..7, halves 8..15 hold K = 16 + (L>>4)*8 + 0..7.
__device__ __forceinline__ v16h load_b32x16(const _Float16* base, int ld, int lane) {
  union { v16h h; uint4 u[2]; } f;
  const _Float16* p = base + (size_t)(lane & 15) * ld + (lane >> 4) * 8;
  f.u[0] = *(const uint4*)(p);
  f.u[1] = *(const uint4*)(p + 16);
  return f.h;
}
__device__ __forceinline__ v8f wmma_f16(v16h a, v16h b, v8f c) {
  return __builtin_amdgcn_wmma_f32_16x16x32_f16(false, a, false, b, (short)0, c, false, false);
}
// C/D 16x16 f32: lane L, reg r -> (m = r + (L>>4)*8, n = L&15)

// async copy of one 16-byte chunk: global -> LDS (ASYNCcnt tracked)
__device__ __forceinline__ void async_g2l_b128(unsigned lds_byte_off, const void* gaddr) {
  asm volatile("global_load_async_to_lds_b128 %0, %1, off"
               :: "v"(lds_byte_off), "v"((unsigned long long)(uintptr_t)gaddr)
               : "memory");
}

// ---------------- kernel 0: fp32 -> fp16 weight prepack ----------------
__global__ void k_wconv(const float* __restrict__ wq, const float* __restrict__ wk,
                        const float* __restrict__ wv, const float* __restrict__ wg,
                        const float* __restrict__ wo,
                        _Float16* __restrict__ Wqkvg, _Float16* __restrict__ Wo) {
  int idx = blockIdx.x * 256 + threadIdx.x;
  if (idx < 512 * 128) {
    int n = idx >> 7;
    float val;
    if      (n < 128) val = wq[idx];
    else if (n < 256) val = wk[idx - 128 * 128];
    else if (n < 384) val = wv[idx - 256 * 128];
    else              val = wg[idx - 384 * 128];
    Wqkvg[idx] = (_Float16)val;
  } else if (idx < 512 * 128 + 128 * 128) {
    int f = idx - 512 * 128;
    Wo[f] = (_Float16)wo[f];
  }
}

// ---------------- kernel 1: LayerNorm + triangle bias ----------------
// 256 threads = 2 rows x 128 channels
__global__ void k_ln_tri(const float* __restrict__ x, const float* __restrict__ ln_w,
                         const float* __restrict__ ln_b, const float* __restrict__ w_tri,
                         _Float16* __restrict__ xn, float* __restrict__ tri) {
  __shared__ float rs1[8], rs2[8], rt[8][4];
  int tid  = threadIdx.x;
  int c    = tid & 127;
  int rsel = tid >> 7;
  int wid  = tid >> 5;
  int lane = tid & 31;
  int row  = blockIdx.x * 2 + rsel;
  float val = x[(size_t)row * C_IN + c];
  float s1 = val, s2 = val * val;
#pragma unroll
  for (int m = 16; m >= 1; m >>= 1) {
    s1 += __shfl_xor(s1, m, 32);
    s2 += __shfl_xor(s2, m, 32);
  }
  if (lane == 0) { rs1[wid] = s1; rs2[wid] = s2; }
  __syncthreads();
  int w0 = rsel * 4;
  float S1 = rs1[w0] + rs1[w0 + 1] + rs1[w0 + 2] + rs1[w0 + 3];
  float S2 = rs2[w0] + rs2[w0 + 1] + rs2[w0 + 2] + rs2[w0 + 3];
  float mu  = S1 * (1.0f / 128.0f);
  float var = S2 * (1.0f / 128.0f) - mu * mu;
  float xv  = (val - mu) * rsqrtf(var + EPS_LN) * ln_w[c] + ln_b[c];
  xn[(size_t)row * C_IN + c] = (_Float16)xv;

  float p[4];
#pragma unroll
  for (int hh = 0; hh < 4; ++hh) {
    float pv = xv * w_tri[hh * C_IN + c];
#pragma unroll
    for (int m = 16; m >= 1; m >>= 1) pv += __shfl_xor(pv, m, 32);
    p[hh] = pv;
  }
  if (lane == 0)
#pragma unroll
    for (int hh = 0; hh < 4; ++hh) rt[wid][hh] = p[hh];
  __syncthreads();
  if (c < 4) {
    float t = rt[w0][c] + rt[w0 + 1][c] + rt[w0 + 2][c] + rt[w0 + 3][c];
    tri[(size_t)c * NROW + row] = t;
  }
}

// ---------------- kernel 2: fused QKV+G projection GEMM ----------------
// M=65536 rows of xn, N=512 (q:0-127, k:128-255, v:256-383, g:384-511), K=128.
// one 16x16 tile per wave, 4x wmma_f32_16x16x32_f16 over K.
__global__ void k_proj(const _Float16* __restrict__ xn, const _Float16* __restrict__ W,
                       const float* __restrict__ b_g,
                       _Float16* __restrict__ q, _Float16* __restrict__ k,
                       _Float16* __restrict__ v, _Float16* __restrict__ og) {
  int lane = threadIdx.x & 31;
  int wave = threadIdx.x >> 5;
  int tile = blockIdx.x * 8 + wave;
  int mt = tile >> 5;   // 4096 M tiles
  int nt = tile & 31;   // 32 N tiles
  v8f acc = {};
#pragma unroll
  for (int kc = 0; kc < 4; ++kc) {
    v16h a = load_a16x32(xn + (size_t)(mt * 16) * C_IN + kc * 32, C_IN, lane);
    v16h b = load_b32x16(W + (size_t)(nt * 16) * C_IN + kc * 32, C_IN, lane);
    acc = wmma_f16(a, b, acc);
  }
  int n     = nt * 16 + (lane & 15);
  int mbase = mt * 16 + ((lane >> 4) * 8);
  if (n < 384) {
    int which = n >> 7;
    int hc = n & 127;
    int hh = hc >> 5, d = hc & 31;
    _Float16* dst = (which == 0) ? q : (which == 1) ? k : v;
    float sc = (which == 0) ? QSCALE : 1.0f;
#pragma unroll
    for (int r = 0; r < 8; ++r) {
      int row = mbase + r;
      int i = row >> 8, j = row & 255;
      dst[(size_t)((i * NH + hh) * J_DIM + j) * HD + d] = (_Float16)(acc[r] * sc);
    }
  } else {
    int cch = n - 384;
    float bg = b_g[cch];
#pragma unroll
    for (int r = 0; r < 8; ++r) {
      int row = mbase + r;
      // fast sigmoid: v_exp_f32 + v_rcp_f32 (avoid IEEE div expansion)
      float gv = __builtin_amdgcn_rcpf(1.0f + __expf(-(acc[r] + bg)));
      og[(size_t)row * C_HID + cch] = (_Float16)gv;
    }
  }
}

// ---------------- kernel 3: fused attention per (i, h) ----------------
// K and V staged into LDS via global_load_async_to_lds_b128 (ASYNCcnt), V then
// transposed LDS->LDS. QK^T: 1 wmma per 16x16 tile (K=HD=32). softmax via
// shfl_xor within 16-lane halves (wave32). P staged in LDS as f16. Fused o*g.
__global__ void k_attn(const _Float16* __restrict__ q, const _Float16* __restrict__ k,
                       const _Float16* __restrict__ v, const float* __restrict__ tri,
                       const float* __restrict__ mask, _Float16* __restrict__ og) {
  extern __shared__ char smem[];
  _Float16* vT    = (_Float16*)(smem + LDS_VT);    // [HD][256]
  _Float16* kS    = (_Float16*)(smem + LDS_KS);    // [256][HD]
  _Float16* vRow  = (_Float16*)(smem + LDS_VROW);  // [256][HD]
  _Float16* pbase = (_Float16*)(smem + LDS_PBASE); // 8 x [16][256]
  // byte offset of the dynamic-LDS region (for async-to-LDS addressing)
  unsigned dynbase = (unsigned)__builtin_amdgcn_groupstaticsize();

  int lane = threadIdx.x & 31;
  int wave = threadIdx.x >> 5;
  int bi   = blockIdx.x >> 2;  // i
  int h    = blockIdx.x & 3;   // head
  size_t base = (size_t)((bi * NH + h) * J_DIM) * HD;
  const _Float16* q_ih = q + base;
  const _Float16* k_ih = k + base;
  const _Float16* v_ih = v + base;

  // async-stage K [256][32] and V [256][32] into LDS: 16 KB each,
  // 256 threads x 4 chunks x 16 B per buffer
#pragma unroll
  for (int it = 0; it < 4; ++it) {
    unsigned off = (unsigned)(it * 4096 + threadIdx.x * 16);
    async_g2l_b128(dynbase + LDS_KS + off, (const char*)k_ih + off);
    async_g2l_b128(dynbase + LDS_VROW + off, (const char*)v_ih + off);
  }
  asm volatile("s_wait_asynccnt 0x0" ::: "memory");
  __syncthreads();

  // transpose V inside LDS: vT[d][key] = vRow[key][d]
  {
    int key = threadIdx.x;
    union { uint4 u[4]; _Float16 h[32]; } vr;
    const uint4* vp = (const uint4*)(vRow + (size_t)key * HD);
#pragma unroll
    for (int c4 = 0; c4 < 4; ++c4) vr.u[c4] = vp[c4];
#pragma unroll
    for (int d = 0; d < HD; ++d) vT[d * J_DIM + key] = vr.h[d];
  }
  __syncthreads();

  _Float16* pstrip = pbase + (size_t)wave * 16 * J_DIM;
  const float* tri_h  = tri + (size_t)h * NROW;
  const float* mask_i = mask + (size_t)bi * J_DIM;
  int nc   = lane & 15;
  int moff = (lane >> 4) * 8;

  for (int t = 0; t < 2; ++t) {
    int qbase = (wave * 2 + t) * 16;
    v16h afrag = load_a16x32(q_ih + (size_t)qbase * HD, HD, lane);
    v8f s[16];
#pragma unroll
    for (int n = 0; n < 16; ++n) {
      v16h b = load_b32x16(kS + (size_t)(n * 16) * HD, HD, lane); // from LDS
      v8f z = {};
      s[n] = wmma_f16(afrag, b, z);
    }
    // add mask bias (depends on key col) + triangle bias (depends on q row, key col)
#pragma unroll
    for (int n = 0; n < 16; ++n) {
      int col = n * 16 + nc;
      float mb = BIGINF * (mask_i[col] - 1.0f);
      const float* tp = tri_h + (size_t)(qbase + moff) * J_DIM + col;
#pragma unroll
      for (int r = 0; r < 8; ++r) s[n][r] += mb + tp[(size_t)r * J_DIM];
    }
    // row softmax over 256 cols: reduce over 16 tiles + 16 lanes in half-wave
#pragma unroll
    for (int r = 0; r < 8; ++r) {
      float mx = -3.0e38f;
#pragma unroll
      for (int n = 0; n < 16; ++n) mx = fmaxf(mx, s[n][r]);
#pragma unroll
      for (int m2 = 8; m2 >= 1; m2 >>= 1) mx = fmaxf(mx, __shfl_xor(mx, m2, 32));
      float sum = 0.0f;
#pragma unroll
      for (int n = 0; n < 16; ++n) { float e = __expf(s[n][r] - mx); s[n][r] = e; sum += e; }
#pragma unroll
      for (int m2 = 8; m2 >= 1; m2 >>= 1) sum += __shfl_xor(sum, m2, 32);
      float inv = __builtin_amdgcn_rcpf(sum); // v_rcp_f32, not IEEE div chain
#pragma unroll
      for (int n = 0; n < 16; ++n) s[n][r] *= inv;
    }
    // spill probs (f16) into this wave's LDS strip: pstrip[m][col]
#pragma unroll
    for (int n = 0; n < 16; ++n) {
      int col = n * 16 + nc;
#pragma unroll
      for (int r = 0; r < 8; ++r)
        pstrip[(moff + r) * J_DIM + col] = (_Float16)s[n][r];
    }
    asm volatile("s_wait_dscnt 0" ::: "memory"); // LDS in-order per wave; drain before re-read

    // o = P(16x256) x V(256x32): 8 K-chunks x 2 N-tiles
    v8f o0 = {}, o1 = {};
#pragma unroll
    for (int kc = 0; kc < 8; ++kc) {
      v16h pa = load_a16x32(pstrip + kc * 32, J_DIM, lane);
      v16h b0 = load_b32x16(vT + kc * 32, J_DIM, lane);
      v16h b1 = load_b32x16(vT + (size_t)16 * J_DIM + kc * 32, J_DIM, lane);
      o0 = wmma_f16(pa, b0, o0);
      o1 = wmma_f16(pa, b1, o1);
    }
    // fused gating: og currently holds g; overwrite with o*g (each addr RMW by one lane)
#pragma unroll
    for (int nb = 0; nb < 2; ++nb) {
      v8f oo = nb ? o1 : o0;
      int d = nb * 16 + nc;
#pragma unroll
      for (int r = 0; r < 8; ++r) {
        int qg = qbase + moff + r;
        size_t addr = (size_t)(bi * J_DIM + qg) * C_HID + h * HD + d;
        float gv = (float)og[addr];
        og[addr] = (_Float16)(oo[r] * gv);
      }
    }
  }
}

// ---------------- kernel 4: output projection GEMM + bias ----------------
__global__ void k_out(const _Float16* __restrict__ og, const _Float16* __restrict__ Wo,
                      const float* __restrict__ b_o, float* __restrict__ out) {
  int lane = threadIdx.x & 31;
  int wave = threadIdx.x >> 5;
  int tile = blockIdx.x * 8 + wave;
  int mt = tile >> 3;  // 4096 M tiles
  int nt = tile & 7;   // 8 N tiles
  v8f acc = {};
#pragma unroll
  for (int kc = 0; kc < 4; ++kc) {
    v16h a = load_a16x32(og + (size_t)(mt * 16) * C_HID + kc * 32, C_HID, lane);
    v16h b = load_b32x16(Wo + (size_t)(nt * 16) * C_IN + kc * 32, C_IN, lane);
    acc = wmma_f16(a, b, acc);
  }
  int cch = nt * 16 + (lane & 15);
  float bo = b_o[cch];
  int mbase = mt * 16 + ((lane >> 4) * 8);
#pragma unroll
  for (int r = 0; r < 8; ++r)
    out[(size_t)(mbase + r) * C_IN + cch] = acc[r] + bo;
}

// ---------------- launcher ----------------
extern "C" void kernel_launch(void* const* d_in, const int* in_sizes, int n_in,
                              void* d_out, int out_size, void* d_ws, size_t ws_size,
                              hipStream_t stream) {
  (void)in_sizes; (void)n_in; (void)out_size; (void)ws_size;
  const float* x     = (const float*)d_in[0];
  const float* mask  = (const float*)d_in[1];
  const float* ln_w  = (const float*)d_in[2];
  const float* ln_b  = (const float*)d_in[3];
  const float* w_tri = (const float*)d_in[4];
  const float* w_q   = (const float*)d_in[5];
  const float* w_k   = (const float*)d_in[6];
  const float* w_v   = (const float*)d_in[7];
  const float* w_g   = (const float*)d_in[8];
  const float* b_g   = (const float*)d_in[9];
  const float* w_o   = (const float*)d_in[10];
  const float* b_o   = (const float*)d_in[11];
  float* out = (float*)d_out;

  char* ws = (char*)d_ws;
  _Float16* xn    = (_Float16*)(ws + OFF_XN);
  _Float16* qb    = (_Float16*)(ws + OFF_Q);
  _Float16* kb    = (_Float16*)(ws + OFF_K);
  _Float16* vb    = (_Float16*)(ws + OFF_V);
  _Float16* og    = (_Float16*)(ws + OFF_OG);
  float*    tri   = (float*)   (ws + OFF_TRI);
  _Float16* Wqkvg = (_Float16*)(ws + OFF_WQKVG);
  _Float16* Wo    = (_Float16*)(ws + OFF_WO);

  k_wconv<<<320, 256, 0, stream>>>(w_q, w_k, w_v, w_g, w_o, Wqkvg, Wo);
  k_ln_tri<<<NROW / 2, 256, 0, stream>>>(x, ln_w, ln_b, w_tri, xn, tri);
  k_proj<<<(4096 * 32) / 8, 256, 0, stream>>>(xn, Wqkvg, b_g, qb, kb, vb, og);
  // dynamic LDS: 112 KB (vT + K + Vrow + 8 prob strips), < 320 KB/WGP
  k_attn<<<I_DIM * NH, 256, LDS_TOTAL, stream>>>(qb, kb, vb, tri, mask, og);
  k_out<<<(4096 * 8) / 8, 256, 0, stream>>>(og, Wo, b_o, out);
}
